// SelfAttention2D_63256278335495
// MI455X (gfx1250) — matrix operand order repaired and checked by reading, not compiled, not run on hardware
//
#include <hip/hip_runtime.h>
#include <hip/hip_bf16.h>

typedef _Float16 f16;
typedef __attribute__((ext_vector_type(16))) _Float16 v16h;
typedef __attribute__((ext_vector_type(8)))  _Float16 v8h;
typedef __attribute__((ext_vector_type(8)))  float    v8f;

#define NB    2
#define CC    256
#define NTOK  4096
#define MROWS 8192      // NB*NTOK
#define NHEAD 8
#define HD    32
#define NGRP  32
// 1/sqrt(32) * log2(e): softmax done in exp2 domain (v_exp_f32 is natively exp2)
#define QSCALE2 (0.17677669529663687f * 1.4426950408889634f)

#define BPAD 272        // LDS row stride in halves: 544B = 17*32B (32B aligned, 8-bank rotate)

// ---------------- WMMA helpers (CDNA5 wave32 layouts, ISA 7.12.2) -------------

__device__ __forceinline__ v8f wmma16(v16h a, v16h b, v8f c) {
  // D = A(16x32 f16) * B(32x16 f16) + C(16x16 f32)
  return __builtin_amdgcn_wmma_f32_16x16x32_f16(false, a, false, b, (short)0, c, false, false);
}

// A-frag: row m = lane%16; elems 0-15 -> k = (lane/16)*16 .. +15 contiguous
__device__ __forceinline__ v16h load_a_frag(const f16* __restrict__ A, int lda) {
  const int lane = threadIdx.x & 31;
  const int m = lane & 15, hi = lane >> 4;
  const f16* p = A + (size_t)m * lda + hi * 16;
  v8h lo8 = *(const v8h*)(p);
  v8h hi8 = *(const v8h*)(p + 8);
  v16h r;
#pragma unroll
  for (int i = 0; i < 8; ++i) { r[i] = lo8[i]; r[i + 8] = hi8[i]; }
  return r;
}

// B-frag from transposed (N-major) storage: col n = lane%16; k = (lane/16)*16 .. +15 contiguous
__device__ __forceinline__ v16h load_b_frag(const f16* __restrict__ BT, int ldb) {
  const int lane = threadIdx.x & 31;
  const int n = lane & 15, hi = lane >> 4;
  return *(const v16h*)(BT + (size_t)n * ldb + hi * 16);
}

// Stage a 64x256 f16 weight strip (32KB) into LDS with CDNA5 async copies.
// Per-lane global_load_async_to_lds_b128: 2048 16B chunks / 256 threads = 8 each.
__device__ __forceinline__ void stage_b_strip_async(const f16* __restrict__ Bbase,
                                                    f16 (*bsh)[BPAD]) {
  const int tid = threadIdx.x;
#pragma unroll
  for (int i = 0; i < 8; ++i) {
    const int c = i * 256 + tid;            // chunk id 0..2047
    const int row = c >> 5;                 // 64 rows
    const int col8 = (c & 31) * 8;          // half offset within row (16B chunks)
    const f16* gp = Bbase + (size_t)row * CC + col8;
    const unsigned ldsa = (unsigned)(unsigned long long)(const void*)&bsh[row][col8];
    asm volatile("global_load_async_to_lds_b128 %0, %1, off"
                 :: "v"(ldsa), "v"(gp) : "memory");
  }
  asm volatile("s_wait_asynccnt 0x0" ::: "memory");
  __syncthreads();
}

// ---------------- GroupNorm -------------------------------------------------

__global__ void gn_stats(const float* __restrict__ x, float* __restrict__ stats) {
  const int bg = blockIdx.x;            // b*32+g, 64 total
  const int b = bg >> 5, g = bg & 31;
  const float* base = x + (size_t)b * NTOK * CC + g * 8;
  float s = 0.f, ss = 0.f;
  for (int n = threadIdx.x; n < NTOK; n += 256) {
    const float* p = base + (size_t)n * CC;
#pragma unroll
    for (int j = 0; j < 8; ++j) { float v = p[j]; s += v; ss += v * v; }
  }
#pragma unroll
  for (int m = 16; m >= 1; m >>= 1) { s += __shfl_xor(s, m, 32); ss += __shfl_xor(ss, m, 32); }
  __shared__ float sh0[8], sh1[8];
  const int wid = threadIdx.x >> 5;
  if ((threadIdx.x & 31) == 0) { sh0[wid] = s; sh1[wid] = ss; }
  __syncthreads();
  if (threadIdx.x == 0) {
    float S = 0.f, SS = 0.f;
#pragma unroll
    for (int i = 0; i < 8; ++i) { S += sh0[i]; SS += sh1[i]; }
    const float mean = S * (1.f / 32768.f);
    const float var  = SS * (1.f / 32768.f) - mean * mean;
    stats[bg * 2]     = mean;
    stats[bg * 2 + 1] = rsqrtf(var + 1e-3f);
  }
}

__global__ void gn_apply(const float* __restrict__ x, const float* __restrict__ gamma,
                         const float* __restrict__ beta, const float* __restrict__ stats,
                         f16* __restrict__ xn) {
  const int idx = blockIdx.x * 256 + threadIdx.x;   // over 8192*256
  const int c = idx & 255;
  const int b = idx >> 20;                          // row = idx>>8; b = row>>12
  const int g = c >> 3;
  const float mean = stats[(b * NGRP + g) * 2];
  const float rstd = stats[(b * NGRP + g) * 2 + 1];
  xn[idx] = (f16)((x[idx] - mean) * rstd * gamma[c] + beta[c]);
}

// ---------------- Weight prep: f32 [K,N] -> f16 transposed [N,K] --------------

__global__ void prep_wT(const float* __restrict__ w, f16* __restrict__ wT, int Ncols) {
  const int j = blockIdx.x;                          // output row (original col)
  for (int kk = threadIdx.x; kk < CC; kk += 256)
    wT[(size_t)j * CC + kk] = (f16)w[(size_t)kk * Ncols + j];
}

// ---------------- QKV GEMM: xn[8192,256] @ w_qkv -> scatter Q/K/Vt ------------
// Block = 8 waves sharing one 64-col weight strip staged to LDS via async copy;
// each wave computes a 16x64 strip. A-frags stream from global (L2 resident).

__global__ void qkv_gemm(const f16* __restrict__ xn, const f16* __restrict__ wqkvT,
                         f16* __restrict__ qbuf, f16* __restrict__ kbuf,
                         f16* __restrict__ vtbuf) {
  const int wid = threadIdx.x >> 5, lane = threadIdx.x & 31;
  const int lo = lane & 15, hi = lane >> 4;
  const int nstrip = blockIdx.x % 12;                // 12 strips of 64 cols
  const int mtile  = (blockIdx.x / 12) * 8 + wid;    // 512 M-tiles of 16 rows
  const f16* Arow  = xn + (size_t)mtile * 16 * CC;
  const f16* Bbase = wqkvT + (size_t)nstrip * 64 * CC;

  __shared__ __align__(32) f16 bsh[64][BPAD];
  stage_b_strip_async(Bbase, bsh);

  v8f acc[4] = {};
#pragma unroll
  for (int k0 = 0; k0 < CC; k0 += 32) {
    v16h a = load_a_frag(Arow + k0, CC);
#pragma unroll
    for (int t = 0; t < 4; ++t) {
      v16h bf = *(const v16h*)&bsh[t * 16 + lo][k0 + hi * 16];
      acc[t] = wmma16(a, bf, acc[t]);
    }
  }

#pragma unroll
  for (int t = 0; t < 4; ++t) {
    const int j = nstrip * 64 + t * 16 + lo;         // 0..767
    const int which = j >> 8;                        // 0=q,1=k,2=v
    const int head = (j & 255) >> 5;
    const int d = j & 31;
#pragma unroll
    for (int r = 0; r < 8; ++r) {
      const int row = mtile * 16 + 8 * hi + r;
      const int b = row >> 12, n = row & 4095;
      const float v = acc[t][r];
      const size_t hidx = (((size_t)b * NHEAD + head) * NTOK + n) * HD + d;
      if (which == 0)      qbuf[hidx] = (f16)(v * QSCALE2);   // exp2-domain logits
      else if (which == 1) kbuf[hidx] = (f16)v;
      else vtbuf[(((size_t)b * NHEAD + head) * HD + d) * NTOK + n] = (f16)v;
    }
  }
}

// ---------------- Flash attention: one wave = 16 query rows -------------------

__global__ void attn_kernel(const f16* __restrict__ q, const f16* __restrict__ k,
                            const f16* __restrict__ vt, f16* __restrict__ aout) {
  const int bh = blockIdx.x >> 5;                    // 16 (b,h) pairs
  const int qchunk = blockIdx.x & 31;                // 32 chunks of 128 rows
  const int wid = threadIdx.x >> 5, lane = threadIdx.x & 31;
  const int lo = lane & 15, hi = lane >> 4;
  const int qtile = qchunk * 8 + wid;                // 16-row tile index

  const f16* Q  = q  + ((size_t)bh * NTOK + qtile * 16) * HD;
  const f16* K  = k  + (size_t)bh * NTOK * HD;
  const f16* VT = vt + (size_t)bh * HD * NTOK;

  __shared__ __align__(32) f16 pbuf[8][16][40];      // per-wave P tile, padded

  const v16h afrag = load_a_frag(Q, HD);             // whole head_dim (K=32) in one frag
  v8f acc0 = {}, acc1 = {};
  float mrow[8], lrow[8];
#pragma unroll
  for (int r = 0; r < 8; ++r) { mrow[r] = -3.0e38f; lrow[r] = 0.f; }

  // prime K-tile fragments for kt = 0
  v16h b0 = load_b_frag(K, HD);
  v16h b1 = load_b_frag(K + (size_t)16 * HD, HD);

  for (int kt = 0; kt < NTOK; kt += 32) {
    const int ktn = (kt + 32 < NTOK) ? (kt + 32) : kt;
    // issue ALL global loads up front: next K tile + current V tile.
    v16h nb0 = load_b_frag(K + (size_t)ktn * HD, HD);
    v16h nb1 = load_b_frag(K + (size_t)(ktn + 16) * HD, HD);
    v16h vb0 = *(const v16h*)(VT + (size_t)lo * NTOK + kt + hi * 16);
    v16h vb1 = *(const v16h*)(VT + (size_t)(16 + lo) * NTOK + kt + hi * 16);
    __builtin_prefetch(VT + (size_t)lo * NTOK + kt + 32, 0, 1);   // global_prefetch_b8

    v8f s0 = {}, s1 = {};
    s0 = wmma16(afrag, b0, s0);
    s1 = wmma16(afrag, b1, s1);

    // online softmax in exp2 domain (scale folded into Q); loads still in flight
    float alpha[8];
#pragma unroll
    for (int r = 0; r < 8; ++r) {
      float tmax = fmaxf(s0[r], s1[r]);
#pragma unroll
      for (int m = 8; m >= 1; m >>= 1) tmax = fmaxf(tmax, __shfl_xor(tmax, m, 32));
      const float mn = fmaxf(mrow[r], tmax);
      const float p0 = exp2f(s0[r] - mn);
      const float p1 = exp2f(s1[r] - mn);
      alpha[r] = exp2f(mrow[r] - mn);
      mrow[r] = mn;
      float ps = p0 + p1;
#pragma unroll
      for (int m = 8; m >= 1; m >>= 1) ps += __shfl_xor(ps, m, 32);
      lrow[r] = lrow[r] * alpha[r] + ps;
      const int row = r + 8 * hi;                    // C-layout row
      pbuf[wid][row][lo]      = (f16)p0;
      pbuf[wid][row][16 + lo] = (f16)p1;
      acc0[r] *= alpha[r];
      acc1[r] *= alpha[r];
    }
    __syncthreads();                                 // uniform: all waves do 128 iters

    // P (C-layout in LDS) -> A-frag
    const f16* pp = &pbuf[wid][lo][0];
    v8h plo = *(const v8h*)(pp + hi * 16);
    v8h phi = *(const v8h*)(pp + hi * 16 + 8);
    v16h pa;
#pragma unroll
    for (int i = 0; i < 8; ++i) { pa[i] = plo[i]; pa[i + 8] = phi[i]; }

    acc0 = wmma16(pa, vb0, acc0);
    acc1 = wmma16(pa, vb1, acc1);
    __syncthreads();                                 // WAR on pbuf
    b0 = nb0;
    b1 = nb1;
  }

  const int b = bh >> 3, h = bh & 7;
#pragma unroll
  for (int r = 0; r < 8; ++r) {
    const float inv = 1.f / lrow[r];
    const int n = qtile * 16 + 8 * hi + r;
    const size_t base = ((size_t)b * NTOK + n) * CC + h * HD;
    aout[base + lo]      = (f16)(acc0[r] * inv);
    aout[base + 16 + lo] = (f16)(acc1[r] * inv);
  }
}

// ---------------- Output projection + bias + residual -------------------------

__global__ void proj_gemm(const f16* __restrict__ ain, const f16* __restrict__ wprojT,
                          const float* __restrict__ bproj, const float* __restrict__ x,
                          float* __restrict__ out) {
  const int wid = threadIdx.x >> 5, lane = threadIdx.x & 31;
  const int lo = lane & 15, hi = lane >> 4;
  const int nstrip = blockIdx.x & 3;                 // 4 strips of 64 cols
  const int mtile  = (blockIdx.x >> 2) * 8 + wid;    // 512 M-tiles
  const f16* Arow  = ain + (size_t)mtile * 16 * CC;
  const f16* Bbase = wprojT + (size_t)nstrip * 64 * CC;

  __shared__ __align__(32) f16 bsh[64][BPAD];
  stage_b_strip_async(Bbase, bsh);

  v8f acc[4] = {};
#pragma unroll
  for (int k0 = 0; k0 < CC; k0 += 32) {
    v16h a = load_a_frag(Arow + k0, CC);
#pragma unroll
    for (int t = 0; t < 4; ++t) {
      v16h bf = *(const v16h*)&bsh[t * 16 + lo][k0 + hi * 16];
      acc[t] = wmma16(a, bf, acc[t]);
    }
  }

#pragma unroll
  for (int t = 0; t < 4; ++t) {
    const int col = nstrip * 64 + t * 16 + lo;
#pragma unroll
    for (int r = 0; r < 8; ++r) {
      const int row = mtile * 16 + 8 * hi + r;
      const size_t idx = (size_t)row * CC + col;
      out[idx] = acc[t][r] + bproj[col] + x[idx];
    }
  }
}

// ---------------- Launch ------------------------------------------------------

extern "C" void kernel_launch(void* const* d_in, const int* in_sizes, int n_in,
                              void* d_out, int out_size, void* d_ws, size_t ws_size,
                              hipStream_t stream) {
  const float* x     = (const float*)d_in[0];
  const float* gamma = (const float*)d_in[1];
  const float* beta  = (const float*)d_in[2];
  const float* wqkv  = (const float*)d_in[3];
  const float* wproj = (const float*)d_in[4];
  const float* bproj = (const float*)d_in[5];
  float* out = (float*)d_out;

  char* ws = (char*)d_ws;
  const size_t OFF_STATS = 0;                                   // 512 B
  const size_t OFF_XN    = 1024;                                // 4 MiB
  const size_t OFF_WQKVT = OFF_XN    + (size_t)MROWS * CC * 2;  // 384 KiB
  const size_t OFF_WPRJT = OFF_WQKVT + (size_t)768 * CC * 2;    // 128 KiB
  const size_t OFF_Q     = OFF_WPRJT + (size_t)CC * CC * 2;     // 4 MiB
  const size_t OFF_K     = OFF_Q  + (size_t)NB * NHEAD * NTOK * HD * 2;
  const size_t OFF_VT    = OFF_K  + (size_t)NB * NHEAD * NTOK * HD * 2;
  const size_t OFF_AOUT  = OFF_VT + (size_t)NB * NHEAD * NTOK * HD * 2;

  float* stats = (float*)(ws + OFF_STATS);
  f16* xn     = (f16*)(ws + OFF_XN);
  f16* wqkvT  = (f16*)(ws + OFF_WQKVT);
  f16* wprjT  = (f16*)(ws + OFF_WPRJT);
  f16* qbuf   = (f16*)(ws + OFF_Q);
  f16* kbuf   = (f16*)(ws + OFF_K);
  f16* vtbuf  = (f16*)(ws + OFF_VT);
  f16* aout   = (f16*)(ws + OFF_AOUT);

  gn_stats<<<NB * NGRP, 256, 0, stream>>>(x, stats);
  gn_apply<<<MROWS, 256, 0, stream>>>(x, gamma, beta, stats, xn);
  prep_wT<<<768, 256, 0, stream>>>(wqkv, wqkvT, 768);
  prep_wT<<<256, 256, 0, stream>>>(wproj, wprjT, 256);
  qkv_gemm<<<768, 256, 0, stream>>>(xn, wqkvT, qbuf, kbuf, vtbuf);
  attn_kernel<<<512, 256, 0, stream>>>(qbuf, kbuf, vtbuf, aout);
  proj_gemm<<<256, 256, 0, stream>>>(aout, wprjT, bproj, x, out);
}
